// Block2DAttention_13709535608873
// MI455X (gfx1250) — matrix operand order repaired and checked by reading, not compiled, not run on hardware
//
#include <hip/hip_runtime.h>
#include <math.h>

typedef __attribute__((ext_vector_type(16))) __bf16 v16bf;
typedef __attribute__((ext_vector_type(8)))  __bf16 v8bf;
typedef __attribute__((ext_vector_type(8)))  float  v8f;
typedef __attribute__((ext_vector_type(4)))  unsigned int u32x4;
typedef __attribute__((ext_vector_type(8)))  int i32x8;
typedef __attribute__((ext_vector_type(4)))  int i32x4;

#if defined(__HIP_DEVICE_COMPILE__) && __has_builtin(__builtin_amdgcn_tensor_load_to_lds)
#define HAVE_TDM 1
#else
#define HAVE_TDM 0
#endif

__device__ __forceinline__ __bf16 f2bf(float f) {
  union { float f; unsigned u; } v; v.f = f;
  unsigned r = v.u + 0x7FFFu + ((v.u >> 16) & 1u);   // round-to-nearest-even
  unsigned short h = (unsigned short)(r >> 16);
  return __builtin_bit_cast(__bf16, h);
}

__device__ __forceinline__ v16bf cat16(v8bf a, v8bf b) {
  return __builtin_shufflevector(a, b, 0,1,2,3,4,5,6,7,8,9,10,11,12,13,14,15);
}

#if HAVE_TDM
__device__ __forceinline__ void tdm_issue(u32x4 g0, i32x8 g1) {
  i32x4 z4 = {0, 0, 0, 0};
#if __clang_major__ >= 23
  i32x8 z8 = {0, 0, 0, 0, 0, 0, 0, 0};
  __builtin_amdgcn_tensor_load_to_lds(g0, g1, z4, z4, z8, 0);
#else
  __builtin_amdgcn_tensor_load_to_lds(g0, g1, z4, z4, 0);
#endif
}

// 2D tile load: tile_d0 elements (bf16, contiguous) x tile_d1 rows, global row
// stride = stride0 elements. pad: +4 dwords (8 elems) after every 16 dwords
// (32 elems) -> padded LDS row stride of 40 elements.
__device__ __forceinline__ void tdm_load_2d(const void* gaddr, unsigned lds_addr,
                                            unsigned tile_d0, unsigned tile_d1,
                                            unsigned long long stride0) {
  unsigned long long ga = (unsigned long long)gaddr;
  u32x4 g0;
  g0[0] = 1u;                                    // count=1 (valid user descriptor)
  g0[1] = lds_addr;                              // LDS byte address
  g0[2] = (unsigned)(ga & 0xFFFFFFFFu);          // global_addr[31:0]
  g0[3] = (unsigned)((ga >> 32) & 0x01FFFFFFu) | (2u << 30);  // addr[56:32] | type=2
  const unsigned td0 = 0x40000000u, td1 = 0x40000000u;        // huge dims: no OOB clip
  i32x8 g1;
  g1[0] = (int)((1u << 16) |                     // data_size = 2 bytes
                (1u << 20) |                     // pad_enable
                (3u << 22) |                     // pad_interval: 16 dwords
                (3u << 25));                     // pad_amount: 4 dwords
  g1[1] = (int)((td0 & 0xFFFFu) << 16);                         // tensor_dim0 lo
  g1[2] = (int)((td0 >> 16) | ((td1 & 0xFFFFu) << 16));         // tdim0 hi | tdim1 lo
  g1[3] = (int)((td1 >> 16) | (tile_d0 << 16));                 // tdim1 hi | tile_dim0
  g1[4] = (int)(tile_d1 & 0xFFFFu);                             // tile_dim1 | tile_dim2=0
  g1[5] = (int)(unsigned)(stride0 & 0xFFFFFFFFull);             // dim0_stride lo32
  g1[6] = (int)(unsigned)((stride0 >> 32) & 0xFFFFull);         // dim0_stride hi16
  g1[7] = 0;
  tdm_issue(g0, g1);
}
#endif

// ---------------------------------------------------------------- converts
__global__ void cvt_bf16(const float* __restrict__ in, __bf16* __restrict__ out, int n) {
  int i = (blockIdx.x * blockDim.x + threadIdx.x) * 8;
  if (i >= n) return;
  float4 a = *(const float4*)(in + i);
  float4 b = *(const float4*)(in + i + 4);
  v8bf o;
  o[0]=f2bf(a.x); o[1]=f2bf(a.y); o[2]=f2bf(a.z); o[3]=f2bf(a.w);
  o[4]=f2bf(b.x); o[5]=f2bf(b.y); o[6]=f2bf(b.z); o[7]=f2bf(b.w);
  *(v8bf*)(out + i) = o;
}

// out[n*K + k] = bf16(in[k*N + n])   (weight transpose for GEMM B-operand)
__global__ void cvt_t_bf16(const float* __restrict__ in, __bf16* __restrict__ out, int K, int N) {
  int idx = blockIdx.x * blockDim.x + threadIdx.x;
  if (idx >= K * N) return;
  int k = idx / N, n = idx - k * N;
  out[(size_t)n * K + k] = f2bf(in[idx]);
}

// ---------------------------------------------------------------- GEMM
// C[M,N] = A[M,K] * Bt[N,K]^T, bf16 in, f32 accumulate. N,K compile-time.
#define BM 128
#define BN 128
#define BK 32
#define LDAB 40  // padded LDS row stride (elements); 80B, 16B-aligned rows

template <int N, int K, bool F32OUT>
__global__ __launch_bounds__(256) void gemm_bf16(
    const __bf16* __restrict__ A, const __bf16* __restrict__ Bt, void* __restrict__ C)
{
  __shared__ __bf16 As[2][BM * LDAB];
  __shared__ __bf16 Bs[2][BN * LDAB];

  const int tid  = threadIdx.x;
  const int lane = tid & 31;
  const int wid  = tid >> 5;
  const int hf   = lane >> 4;
  const int ln   = lane & 15;
  const int wm   = wid >> 1;  // 0..3 (M tiles of 32)
  const int wn   = wid & 1;   // 0..1 (N tiles of 64)
  const int bm   = blockIdx.y * BM;
  const int bn   = blockIdx.x * BN;

  v8f zero = {0,0,0,0,0,0,0,0};
  v8f acc[2][4];
  for (int i = 0; i < 2; i++) for (int j = 0; j < 4; j++) acc[i][j] = zero;

  const int nk = K / BK;

#if HAVE_TDM
  // ---- Tensor Data Mover staging, double buffered: TDM prefetches tile kk+1
  // while 8 waves run WMMAs on tile kk. Wave 0 issues; s_wait_tensorcnt gates.
  const __bf16* Ablk = A  + (size_t)bm * K;
  const __bf16* Bblk = Bt + (size_t)bn * K;
  const unsigned aLds = (unsigned)(size_t)&As[0][0];
  const unsigned bLds = (unsigned)(size_t)&Bs[0][0];
  if (wid == 0) {
    tdm_load_2d(Ablk, aLds, BK, BM, K);
    tdm_load_2d(Bblk, bLds, BK, BN, K);
  }
  for (int kk = 0; kk < nk; kk++) {
    const int p = kk & 1;
    if (wid == 0) {
      if (kk + 1 < nk) {
        tdm_load_2d(Ablk + (size_t)(kk + 1) * BK, aLds + (unsigned)(p ^ 1) * BM * LDAB * 2, BK, BM, K);
        tdm_load_2d(Bblk + (size_t)(kk + 1) * BK, bLds + (unsigned)(p ^ 1) * BN * LDAB * 2, BK, BN, K);
        __builtin_amdgcn_s_wait_tensorcnt(2);   // tile p complete; next tile in flight
      } else {
        __builtin_amdgcn_s_wait_tensorcnt(0);
      }
    }
    __syncthreads();
    v16bf af[2], bfr[4];
#pragma unroll
    for (int ms = 0; ms < 2; ms++) {
      const __bf16* p2 = &As[p][(wm * 32 + ms * 16 + ln) * LDAB];
      af[ms] = cat16(*(const v8bf*)(p2 + 8 * hf), *(const v8bf*)(p2 + 16 + 8 * hf));
    }
#pragma unroll
    for (int ns = 0; ns < 4; ns++) {
      const __bf16* p2 = &Bs[p][(wn * 64 + ns * 16 + ln) * LDAB];
      bfr[ns] = cat16(*(const v8bf*)(p2 + 8 * hf), *(const v8bf*)(p2 + 16 + 8 * hf));
    }
#pragma unroll
    for (int ms = 0; ms < 2; ms++)
#pragma unroll
      for (int ns = 0; ns < 4; ns++)
        acc[ms][ns] = __builtin_amdgcn_wmma_f32_16x16x32_bf16(
            false, af[ms], false, bfr[ns], (short)0, acc[ms][ns], false, false);
    __syncthreads();   // all waves done with buffer p before TDM overwrites it
  }
#else
  // ---- fallback: cooperative vmem->ds staging (single buffer)
  const int sr = tid >> 1;         // 0..127
  const int sc = (tid & 1) * 16;   // 0 / 16
  for (int kk = 0; kk < nk; kk++) {
    const int k0 = kk * BK;
    __syncthreads();
    {
      const __bf16* ga = A + (size_t)(bm + sr) * K + k0 + sc;
      *(v8bf*)(&As[0][sr * LDAB + sc])     = *(const v8bf*)(ga);
      *(v8bf*)(&As[0][sr * LDAB + sc + 8]) = *(const v8bf*)(ga + 8);
      const __bf16* gb = Bt + (size_t)(bn + sr) * K + k0 + sc;
      *(v8bf*)(&Bs[0][sr * LDAB + sc])     = *(const v8bf*)(gb);
      *(v8bf*)(&Bs[0][sr * LDAB + sc + 8]) = *(const v8bf*)(gb + 8);
    }
    __syncthreads();
    v16bf af[2], bfr[4];
#pragma unroll
    for (int ms = 0; ms < 2; ms++) {
      const __bf16* p2 = &As[0][(wm * 32 + ms * 16 + ln) * LDAB];
      af[ms] = cat16(*(const v8bf*)(p2 + 8 * hf), *(const v8bf*)(p2 + 16 + 8 * hf));
    }
#pragma unroll
    for (int ns = 0; ns < 4; ns++) {
      const __bf16* p2 = &Bs[0][(wn * 64 + ns * 16 + ln) * LDAB];
      bfr[ns] = cat16(*(const v8bf*)(p2 + 8 * hf), *(const v8bf*)(p2 + 16 + 8 * hf));
    }
#pragma unroll
    for (int ms = 0; ms < 2; ms++)
#pragma unroll
      for (int ns = 0; ns < 4; ns++)
        acc[ms][ns] = __builtin_amdgcn_wmma_f32_16x16x32_bf16(
            false, af[ms], false, bfr[ns], (short)0, acc[ms][ns], false, false);
  }
#endif

  // Epilogue: single base address, compile-time offsets.
  const size_t base = (size_t)(bm + wm * 32 + 8 * hf) * N + (bn + wn * 64 + ln);
  if constexpr (F32OUT) {
    float* Cp = (float*)C + base;
#pragma unroll
    for (int ms = 0; ms < 2; ms++)
#pragma unroll
      for (int ns = 0; ns < 4; ns++)
#pragma unroll
        for (int i = 0; i < 8; i++)
          Cp[(size_t)(ms * 16 + i) * N + ns * 16] = acc[ms][ns][i];
  } else {
    __bf16* Cp = (__bf16*)C + base;
#pragma unroll
    for (int ms = 0; ms < 2; ms++)
#pragma unroll
      for (int ns = 0; ns < 4; ns++)
#pragma unroll
        for (int i = 0; i < 8; i++)
          Cp[(size_t)(ms * 16 + i) * N + ns * 16] = f2bf(acc[ms][ns][i]);
  }
}

// ---------------------------------------------------------------- attention
// One WG per (block nb, head h, 128 q-rows). 8 waves x 16 q-rows.
#define QT  128
#define KC  64
#define KLD 72   // padded LDS row stride
#define PLD 72

__global__ __launch_bounds__(256) void attn_kernel(
    const __bf16* __restrict__ Q, const __bf16* __restrict__ Km,
    const __bf16* __restrict__ Vm, __bf16* __restrict__ O)
{
  __shared__ __bf16 Ks[KC * KLD];      // [key][d]
  __shared__ __bf16 Vt[64 * KLD];      // [d][key]  (transposed for PV B-operand)
  __shared__ __bf16 Ps[8 * 16 * PLD];  // per-wave P tile [16][PLD]

  const int tid  = threadIdx.x;
  const int lane = tid & 31, wid = tid >> 5;
  const int hf   = lane >> 4, ln = lane & 15;
  const int qt = blockIdx.x, h = blockIdx.y, nb = blockIdx.z;
  const int hk = h >> 2;                          // GQA: 4 q-heads per kv-head
  const int b = nb >> 2, bx = (nb >> 1) & 1, by = nb & 1;
  const int rb = b * 4096 + bx * 2048 + by * 32;  // row(t) = rb + (t>>5)*64 + (t&31)

  // Q fragments live in registers for the whole kernel (d=64 -> 2 k-steps)
  v16bf qf[2];
  {
    int t = qt * QT + wid * 16 + ln;
    size_t qrow = (size_t)(rb + ((t >> 5) << 6) + (t & 31));
    const __bf16* qp = Q + qrow * 2048 + h * 64;
    qf[0] = cat16(*(const v8bf*)(qp + 8 * hf),      *(const v8bf*)(qp + 16 + 8 * hf));
    qf[1] = cat16(*(const v8bf*)(qp + 32 + 8 * hf), *(const v8bf*)(qp + 48 + 8 * hf));
  }

  v8f zero = {0,0,0,0,0,0,0,0};
  v8f o_acc[4]; for (int i = 0; i < 4; i++) o_acc[i] = zero;
  float mrow[8], lrow[8];
  for (int i = 0; i < 8; i++) { mrow[i] = -1e30f; lrow[i] = 0.0f; }

  const int sr = tid >> 2;          // 0..63 (key row)
  const int sc = (tid & 3) * 16;    // d chunk
  __bf16* Pw = &Ps[wid * 16 * PLD];

  for (int c = 0; c < 1024 / KC; c++) {
    __syncthreads();
    {
      int t = c * KC + sr;
      size_t krow = (size_t)(rb + ((t >> 5) << 6) + (t & 31));
      const __bf16* kp = Km + krow * 512 + hk * 64 + sc;
      *(v8bf*)(&Ks[sr * KLD + sc])     = *(const v8bf*)(kp);
      *(v8bf*)(&Ks[sr * KLD + sc + 8]) = *(const v8bf*)(kp + 8);
      const __bf16* vp = Vm + krow * 512 + hk * 64 + sc;
      v8bf v0 = *(const v8bf*)(vp);
      v8bf v1 = *(const v8bf*)(vp + 8);
#pragma unroll
      for (int e = 0; e < 8; e++) {
        Vt[(sc + e) * KLD + sr]     = v0[e];
        Vt[(sc + 8 + e) * KLD + sr] = v1[e];
      }
    }
    __syncthreads();

    // S = Q K^T  (4 n-subtiles x 2 k-steps)
    v8f s[4]; for (int i = 0; i < 4; i++) s[i] = zero;
#pragma unroll
    for (int sn = 0; sn < 4; sn++) {
      const __bf16* kb = &Ks[(sn * 16 + ln) * KLD];
#pragma unroll
      for (int f = 0; f < 2; f++) {
        const __bf16* p = kb + f * 32;
        v16bf bfr = cat16(*(const v8bf*)(p + 8 * hf), *(const v8bf*)(p + 16 + 8 * hf));
        s[sn] = __builtin_amdgcn_wmma_f32_16x16x32_bf16(
            false, qf[f], false, bfr, (short)0, s[sn], false, false);
      }
    }

    // online softmax: reg i <-> row (i + 8*hf); N across lanes 0..15 of the half
#pragma unroll
    for (int i = 0; i < 8; i++) {
      float v0 = fmaxf(fmaxf(s[0][i], s[1][i]), fmaxf(s[2][i], s[3][i])) * 0.125f;
      for (int m = 1; m <= 8; m <<= 1) v0 = fmaxf(v0, __shfl_xor(v0, m, 32));
      float mn = fmaxf(mrow[i], v0);
      float alpha = __expf(mrow[i] - mn);
      mrow[i] = mn;
      float rs = 0.0f;
#pragma unroll
      for (int sn = 0; sn < 4; sn++) {
        float pv = __expf(s[sn][i] * 0.125f - mn);
        s[sn][i] = pv;
        rs += pv;
      }
      for (int m = 1; m <= 8; m <<= 1) rs += __shfl_xor(rs, m, 32);
      lrow[i] = lrow[i] * alpha + rs;
#pragma unroll
      for (int ns = 0; ns < 4; ns++) o_acc[ns][i] *= alpha;
    }

    // C-layout P -> wave-private LDS (re-shape to A-fragment layout)
#pragma unroll
    for (int sn = 0; sn < 4; sn++)
#pragma unroll
      for (int i = 0; i < 8; i++)
        Pw[(i + 8 * hf) * PLD + sn * 16 + ln] = f2bf(s[sn][i]);

    // O += P V  (2 k-steps over 64 keys x 4 d-subtiles)
#pragma unroll
    for (int f = 0; f < 2; f++) {
      const __bf16* pp = Pw + ln * PLD + f * 32;
      v16bf pf = cat16(*(const v8bf*)(pp + 8 * hf), *(const v8bf*)(pp + 16 + 8 * hf));
#pragma unroll
      for (int ns = 0; ns < 4; ns++) {
        const __bf16* vb = &Vt[(ns * 16 + ln) * KLD] + f * 32;
        v16bf vf = cat16(*(const v8bf*)(vb + 8 * hf), *(const v8bf*)(vb + 16 + 8 * hf));
        o_acc[ns] = __builtin_amdgcn_wmma_f32_16x16x32_bf16(
            false, pf, false, vf, (short)0, o_acc[ns], false, false);
      }
    }
  }

  // write O (plain reshape: row = nb*1024 + t)
  {
    size_t base = ((size_t)nb * 1024 + qt * QT + wid * 16 + 8 * hf) * 2048 + h * 64 + ln;
    __bf16* Op = O + base;
#pragma unroll
    for (int i = 0; i < 8; i++) {
      float inv = 1.0f / lrow[i];
#pragma unroll
      for (int ns = 0; ns < 4; ns++)
        Op[(size_t)i * 2048 + ns * 16] = f2bf(o_acc[ns][i] * inv);
    }
  }
}

// ---------------------------------------------------------------- launch
extern "C" void kernel_launch(void* const* d_in, const int* in_sizes, int n_in,
                              void* d_out, int out_size, void* d_ws, size_t ws_size,
                              hipStream_t stream) {
  const float* hs = (const float*)d_in[0];
  const float* Wq = (const float*)d_in[1];
  const float* Wk = (const float*)d_in[2];
  const float* Wv = (const float*)d_in[3];
  const float* Wo = (const float*)d_in[4];
  float* out = (float*)d_out;

  char* ws = (char*)d_ws;
  size_t off = 0;
  auto alloc = [&](size_t bytes) -> void* {
    void* p = ws + off; off += (bytes + 255) & ~(size_t)255; return p;
  };
  __bf16* HS16 = (__bf16*)alloc(8192ull * 2048 * 2);
  __bf16* WqT  = (__bf16*)alloc(2048ull * 2048 * 2);
  __bf16* WkT  = (__bf16*)alloc(512ull  * 2048 * 2);
  __bf16* WvT  = (__bf16*)alloc(512ull  * 2048 * 2);
  __bf16* WoT  = (__bf16*)alloc(2048ull * 2048 * 2);
  __bf16* Q16  = (__bf16*)alloc(8192ull * 2048 * 2);
  __bf16* K16  = (__bf16*)alloc(8192ull * 512  * 2);
  __bf16* V16  = (__bf16*)alloc(8192ull * 512  * 2);
  __bf16* O16  = (__bf16*)alloc(8192ull * 2048 * 2);

  cvt_bf16<<<(16777216 / 8) / 256, 256, 0, stream>>>(hs, HS16, 16777216);
  cvt_t_bf16<<<(2048 * 2048) / 256, 256, 0, stream>>>(Wq, WqT, 2048, 2048);
  cvt_t_bf16<<<(2048 * 512)  / 256, 256, 0, stream>>>(Wk, WkT, 2048, 512);
  cvt_t_bf16<<<(2048 * 512)  / 256, 256, 0, stream>>>(Wv, WvT, 2048, 512);
  cvt_t_bf16<<<(2048 * 2048) / 256, 256, 0, stream>>>(Wo, WoT, 2048, 2048);

  gemm_bf16<2048, 2048, false><<<dim3(2048 / BN, 8192 / BM), 256, 0, stream>>>(HS16, WqT, Q16);
  gemm_bf16<512,  2048, false><<<dim3(512  / BN, 8192 / BM), 256, 0, stream>>>(HS16, WkT, K16);
  gemm_bf16<512,  2048, false><<<dim3(512  / BN, 8192 / BM), 256, 0, stream>>>(HS16, WvT, V16);

  attn_kernel<<<dim3(8, 32, 8), 256, 0, stream>>>(Q16, K16, V16, O16);

  gemm_bf16<2048, 2048, true><<<dim3(2048 / BN, 8192 / BM), 256, 0, stream>>>(O16, WoT, out);
}
